// Pocket_GNN_52055003627593
// MI455X (gfx1250) — compile-verified, matrix operand order checked by reading, not run on hardware
//
#include <hip/hip_runtime.h>
#include <hip/hip_bf16.h>
#include <math.h>

#define NNODES 50000
#define NEDGES 800000
#define NHEADS 8

typedef __attribute__((ext_vector_type(16))) __bf16 v16bf;
typedef __attribute__((ext_vector_type(8)))  float  v8f;

__device__ __forceinline__ float elu1(float v) {
  return v > 0.f ? v : __expf(v) - 1.f;
}

// Order-preserving float <-> uint encoding for atomicMax on floats.
__device__ __forceinline__ unsigned ord_enc(float f) {
  unsigned b = __float_as_uint(f);
  return (b & 0x80000000u) ? ~b : (b | 0x80000000u);
}
__device__ __forceinline__ float ord_dec(unsigned u) {
  return (u & 0x80000000u) ? __uint_as_float(u & 0x7fffffffu)
                           : __uint_as_float(~u);
}

// Load one lane's 16 bf16 values for a K=32 chunk: elements 0..7 come from
// p[0..7], elements 8..15 from p[16..23] (per the 16-bit A/B VGPR layout).
// All offsets are 16-byte aligned by construction.
__device__ __forceinline__ v16bf load_chunk_bf16(const float* __restrict__ p) {
  const float4 f0 = *(const float4*)(p);
  const float4 f1 = *(const float4*)(p + 4);
  const float4 f2 = *(const float4*)(p + 16);
  const float4 f3 = *(const float4*)(p + 20);
  v16bf r;
  r[0]  = (__bf16)f0.x; r[1]  = (__bf16)f0.y; r[2]  = (__bf16)f0.z; r[3]  = (__bf16)f0.w;
  r[4]  = (__bf16)f1.x; r[5]  = (__bf16)f1.y; r[6]  = (__bf16)f1.z; r[7]  = (__bf16)f1.w;
  r[8]  = (__bf16)f2.x; r[9]  = (__bf16)f2.y; r[10] = (__bf16)f2.z; r[11] = (__bf16)f2.w;
  r[12] = (__bf16)f3.x; r[13] = (__bf16)f3.y; r[14] = (__bf16)f3.z; r[15] = (__bf16)f3.w;
  return r;
}

// ---------------------------------------------------------------------------
// Y[M x NOUT] = X[M x K] @ W^T + bias.  W is [NOUT x K] row-major fp32.
// K, NOUT compile-time; M multiple of 16.
// One wave computes a 16x64 output strip (4 n-tiles), A held in registers.
// All W-loads / Y-stores use one base pointer + immediate offsets.
// ---------------------------------------------------------------------------
template <int K, int NOUT>
__global__ void gemm_bias_wmma(const float* __restrict__ X,
                               const float* __restrict__ W,
                               const float* __restrict__ B,
                               float* __restrict__ Y,
                               int M) {
  constexpr int KC = K / 32;   // K chunks
  const int lane = threadIdx.x & 31;
  const int wib  = threadIdx.x >> 5;
  const int wpb  = blockDim.x >> 5;
  const int wave = blockIdx.x * wpb + wib;
  constexpr int NGROUPS = NOUT / 64;
  const int mtiles = M >> 4;
  if (wave >= mtiles * NGROUPS) return;
  const int mt = wave / NGROUPS;
  const int ng = wave - mt * NGROUPS;
  const int half = lane >> 4;            // 0 or 1
  const int l16  = lane & 15;
  const int arow = mt * 16 + l16;        // M = 50000 = 3125*16: no guard
  const int koff = half * 8;

  // A chunks loaded once, reused for all 4 n-tiles.
  v16bf a[KC];
  {
    const float* ap = X + arow * K + koff;
#pragma unroll
    for (int c = 0; c < KC; ++c) a[c] = load_chunk_bf16(ap + c * 32);
  }

  const int bcol0 = ng * 64 + l16;
  const float* bp0 = W + bcol0 * K + koff;
  float* yp0 = Y + (mt * 16 + half * 8) * NOUT + bcol0;

#pragma unroll
  for (int t = 0; t < 4; ++t) {
    v8f acc = {};
#pragma unroll
    for (int c = 0; c < KC; ++c) {
      v16bf b = load_chunk_bf16(bp0 + t * 16 * K + c * 32);
      acc = __builtin_amdgcn_wmma_f32_16x16x32_bf16(false, a[c], false, b,
                                                    (short)0, acc, false, false);
    }
    const float bias = B[bcol0 + t * 16];
#pragma unroll
    for (int j = 0; j < 8; ++j) {
      yp0[j * NOUT + t * 16] = acc[j] + bias;   // immediate offsets
    }
  }
}

// ---------------------------------------------------------------------------
// Input MLP: two fused 23->23 linears with ELU; output padded to stride 32.
// ---------------------------------------------------------------------------
__global__ void input_mlp_kernel(const float* __restrict__ x,
                                 const float* __restrict__ WA,
                                 const float* __restrict__ bA,
                                 const float* __restrict__ WB,
                                 const float* __restrict__ bB,
                                 float* __restrict__ h0, int Nn) {
  int n = blockIdx.x * blockDim.x + threadIdx.x;
  if (n >= Nn) return;
  float xi[23], t[23];
#pragma unroll
  for (int j = 0; j < 23; ++j) xi[j] = x[n * 23 + j];
#pragma unroll 1
  for (int o = 0; o < 23; ++o) {
    float a = bA[o];
#pragma unroll
    for (int j = 0; j < 23; ++j) a = fmaf(xi[j], WA[o * 23 + j], a);
    t[o] = elu1(a);
  }
#pragma unroll 1
  for (int o = 0; o < 23; ++o) {
    float a = bB[o];
#pragma unroll
    for (int j = 0; j < 23; ++j) a = fmaf(t[j], WB[o * 23 + j], a);
    h0[n * 32 + o] = elu1(a);
  }
#pragma unroll
  for (int o = 23; o < 32; ++o) h0[n * 32 + o] = 0.f;  // K padding
}

// Pad layer-1 weights [192 x 23] -> [192 x 32] (zero-filled).
__global__ void pad_w_kernel(const float* __restrict__ Wl,
                             const float* __restrict__ Wr,
                             float* __restrict__ wlp,
                             float* __restrict__ wrp) {
  int i = blockIdx.x * blockDim.x + threadIdx.x;
  if (i >= 192 * 32) return;
  int o = i >> 5, j = i & 31;
  wlp[i] = (j < 23) ? Wl[o * 23 + j] : 0.f;
  wrp[i] = (j < 23) ? Wr[o * 23 + j] : 0.f;
}

// ---------------------------------------------------------------------------
// Edge pass 1: one thread per (edge, head).
// alpha[e,h] = sum_c leakyrelu(xl[s,h,c] + xr[d,h,c] + ea@We[h,c,:]) * att[h,c]
// atomicMax amax[d,h] (ordered-uint encoding).
// ---------------------------------------------------------------------------
template <int C>
__global__ void edge_alpha_kernel(const float* __restrict__ xl,
                                  const float* __restrict__ xr,
                                  const float* __restrict__ eattr,
                                  const int* __restrict__ src,
                                  const int* __restrict__ dst,
                                  const float* __restrict__ We,
                                  const float* __restrict__ att,
                                  float* __restrict__ alpha,
                                  unsigned* __restrict__ amax,
                                  int E) {
  constexpr int HC = NHEADS * C;
  int idx = blockIdx.x * blockDim.x + threadIdx.x;
  if (idx >= E * NHEADS) return;
  int e = idx >> 3;
  int h = idx & 7;
  int s = src[e], d = dst[e];
  float ea[7];
#pragma unroll
  for (int j = 0; j < 7; ++j) ea[j] = eattr[e * 7 + j];
  const float4* pl = (const float4*)(xl + s * HC + h * C);
  const float4* pr = (const float4*)(xr + d * HC + h * C);
  float msum[C];
#pragma unroll
  for (int q = 0; q < C / 4; ++q) {
    float4 va = pl[q];
    float4 vb = pr[q];
    msum[4 * q + 0] = va.x + vb.x;
    msum[4 * q + 1] = va.y + vb.y;
    msum[4 * q + 2] = va.z + vb.z;
    msum[4 * q + 3] = va.w + vb.w;
  }
  const float* wrow = We + h * C * 7;
  const float* arow = att + h * C;
  float acc = 0.f;
#pragma unroll
  for (int c = 0; c < C; ++c) {
    float m = msum[c];
    const float* w = wrow + c * 7;
#pragma unroll
    for (int j = 0; j < 7; ++j) m = fmaf(ea[j], w[j], m);
    m = m > 0.f ? m : 0.2f * m;          // LeakyReLU(0.2)
    acc = fmaf(m, arow[c], acc);
  }
  alpha[idx] = acc;
  atomicMax(&amax[d * NHEADS + h], ord_enc(acc));
}

// ---------------------------------------------------------------------------
// Edge pass 2: ex = exp(alpha - amax[dst]); den[dst,h] += ex;
// out[dst,h,:] += ex * xl[src,h,:].  (Normalization folded into node pass.)
// ---------------------------------------------------------------------------
template <int C>
__global__ void edge_accum_kernel(const float* __restrict__ xl,
                                  const int* __restrict__ src,
                                  const int* __restrict__ dst,
                                  const float* __restrict__ alpha,
                                  const unsigned* __restrict__ amax,
                                  float* __restrict__ den,
                                  float* __restrict__ out,
                                  int E) {
  constexpr int HC = NHEADS * C;
  int idx = blockIdx.x * blockDim.x + threadIdx.x;
  if (idx >= E * NHEADS) return;
  int e = idx >> 3;
  int h = idx & 7;
  int s = src[e], d = dst[e];
  float am = ord_dec(amax[d * NHEADS + h]);
  float ex = __expf(alpha[idx] - am);
  atomicAdd(&den[d * NHEADS + h], ex);
  const float4* pl = (const float4*)(xl + s * HC + h * C);
  float* o = out + d * HC + h * C;
#pragma unroll
  for (int q = 0; q < C / 4; ++q) {
    float4 v = pl[q];
    atomicAdd(&o[4 * q + 0], ex * v.x);
    atomicAdd(&o[4 * q + 1], ex * v.y);
    atomicAdd(&o[4 * q + 2], ex * v.z);
    atomicAdd(&o[4 * q + 3], ex * v.w);
  }
}

// ---------------------------------------------------------------------------
// Node pass: out = out / (den + 1e-16) + bias, then ELU (in place).
// ---------------------------------------------------------------------------
template <int C>
__global__ void node_finalize_kernel(float* __restrict__ out,
                                     const float* __restrict__ den,
                                     const float* __restrict__ bias,
                                     int Nn) {
  constexpr int HC = NHEADS * C;
  int idx = blockIdx.x * blockDim.x + threadIdx.x;
  if (idx >= Nn * HC) return;
  int n = idx / HC;
  int o = idx - n * HC;
  int h = o / C;
  float v = out[idx] / (den[n * NHEADS + h] + 1e-16f) + bias[o];
  out[idx] = elu1(v);
}

// ---------------------------------------------------------------------------
// Head: lin1 (64->20) + ELU, lin2 (20->1).
// ---------------------------------------------------------------------------
__global__ void head_kernel(const float* __restrict__ h,
                            const float* __restrict__ W1,
                            const float* __restrict__ b1,
                            const float* __restrict__ W2,
                            const float* __restrict__ b2,
                            float* __restrict__ out, int Nn) {
  int n = blockIdx.x * blockDim.x + threadIdx.x;
  if (n >= Nn) return;
  float hv[64];
#pragma unroll
  for (int j = 0; j < 64; ++j) hv[j] = h[n * 64 + j];
  float acc2 = b2[0];
#pragma unroll 1
  for (int o = 0; o < 20; ++o) {
    float a = b1[o];
#pragma unroll
    for (int j = 0; j < 64; ++j) a = fmaf(hv[j], W1[o * 64 + j], a);
    acc2 = fmaf(elu1(a), W2[o], acc2);
  }
  out[n] = acc2;
}

// ---------------------------------------------------------------------------
// Host launcher.
// d_in layout (recursive dict-insertion-order flatten of setup_inputs()):
//  0:x  1:edge_index  2:edge_attr  3:linA_w 4:linA_b 5:linB_w 6:linB_b
//  7 + 7*(k-1): c{k}_Wl, c{k}_bl, c{k}_Wr, c{k}_br, c{k}_We, c{k}_att, c{k}_bias
//  35:lin1_w 36:lin1_b 37:lin2_w 38:lin2_b
// ---------------------------------------------------------------------------
extern "C" void kernel_launch(void* const* d_in, const int* in_sizes, int n_in,
                              void* d_out, int out_size, void* d_ws, size_t ws_size,
                              hipStream_t stream) {
  const float* x      = (const float*)d_in[0];
  const int*   eindex = (const int*)  d_in[1];
  const float* eattr  = (const float*)d_in[2];
  const float* linA_w = (const float*)d_in[3];
  const float* linA_b = (const float*)d_in[4];
  const float* linB_w = (const float*)d_in[5];
  const float* linB_b = (const float*)d_in[6];
  const float* lin1_w = (const float*)d_in[35];
  const float* lin1_b = (const float*)d_in[36];
  const float* lin2_w = (const float*)d_in[37];
  const float* lin2_b = (const float*)d_in[38];
  const int* src = eindex;            // edge_index[0]
  const int* dst = eindex + NEDGES;   // edge_index[1]

  // Workspace carve-up (~183 MB total; all sub-buffers 16B-aligned).
  float* bufA  = (float*)d_ws;                          // N*192 (also N*32 padded h0)
  float* bufB  = bufA + (size_t)NNODES * 192;           // N*192
  float* xl    = bufB + (size_t)NNODES * 192;           // N*192
  float* xr    = xl   + (size_t)NNODES * 192;           // N*192
  float* alpha = xr   + (size_t)NNODES * 192;           // E*8
  unsigned* amax = (unsigned*)(alpha + (size_t)NEDGES * NHEADS);  // N*8
  float* den   = (float*)(amax + (size_t)NNODES * NHEADS);        // N*8
  float* wlpad = den + (size_t)NNODES * NHEADS;         // 192*32
  float* wrpad = wlpad + 192 * 32;                      // 192*32

  // Input MLP -> bufA (N x 32, zero-padded cols 23..31).
  input_mlp_kernel<<<(NNODES + 255) / 256, 256, 0, stream>>>(
      x, linA_w, linA_b, linB_w, linB_b, bufA, NNODES);

  const int CArr[4] = {24, 24, 24, 8};
  float* hin  = bufA;
  float* hout = bufB;
  for (int k = 0; k < 4; ++k) {
    const int pb = 7 + k * 7;
    const float* Wl   = (const float*)d_in[pb + 0];
    const float* bl   = (const float*)d_in[pb + 1];
    const float* Wr   = (const float*)d_in[pb + 2];
    const float* br   = (const float*)d_in[pb + 3];
    const float* We   = (const float*)d_in[pb + 4];
    const float* att  = (const float*)d_in[pb + 5];
    const float* bias = (const float*)d_in[pb + 6];
    const int C  = CArr[k];
    const int HC = NHEADS * C;   // 192 or 64

    // Node transforms (bf16 WMMA GEMMs). One wave per 16x64 strip.
    const int mtiles = NNODES / 16;          // exact
    const int waves  = mtiles * (HC / 64);
    const int gblocks = (waves + 7) / 8;     // 8 waves per 256-thread block
    if (k == 0) {
      pad_w_kernel<<<(192 * 32 + 255) / 256, 256, 0, stream>>>(Wl, Wr, wlpad, wrpad);
      gemm_bias_wmma<32, 192><<<gblocks, 256, 0, stream>>>(hin, wlpad, bl, xl, NNODES);
      gemm_bias_wmma<32, 192><<<gblocks, 256, 0, stream>>>(hin, wrpad, br, xr, NNODES);
    } else if (HC == 192) {
      gemm_bias_wmma<192, 192><<<gblocks, 256, 0, stream>>>(hin, Wl, bl, xl, NNODES);
      gemm_bias_wmma<192, 192><<<gblocks, 256, 0, stream>>>(hin, Wr, br, xr, NNODES);
    } else {
      gemm_bias_wmma<192, 64><<<gblocks, 256, 0, stream>>>(hin, Wl, bl, xl, NNODES);
      gemm_bias_wmma<192, 64><<<gblocks, 256, 0, stream>>>(hin, Wr, br, xr, NNODES);
    }

    hipMemsetAsync(amax, 0, sizeof(unsigned) * (size_t)NNODES * NHEADS, stream);
    hipMemsetAsync(den,  0, sizeof(float)    * (size_t)NNODES * NHEADS, stream);
    hipMemsetAsync(hout, 0, sizeof(float)    * (size_t)NNODES * HC, stream);

    const int total_eh = NEDGES * NHEADS;
    const int eblocks  = (total_eh + 255) / 256;
    const int nthreads = NNODES * HC;
    const int nblocks  = (nthreads + 255) / 256;
    if (C == 24) {
      edge_alpha_kernel<24><<<eblocks, 256, 0, stream>>>(
          xl, xr, eattr, src, dst, We, att, alpha, amax, NEDGES);
      edge_accum_kernel<24><<<eblocks, 256, 0, stream>>>(
          xl, src, dst, alpha, amax, den, hout, NEDGES);
      node_finalize_kernel<24><<<nblocks, 256, 0, stream>>>(hout, den, bias, NNODES);
    } else {
      edge_alpha_kernel<8><<<eblocks, 256, 0, stream>>>(
          xl, xr, eattr, src, dst, We, att, alpha, amax, NEDGES);
      edge_accum_kernel<8><<<eblocks, 256, 0, stream>>>(
          xl, src, dst, alpha, amax, den, hout, NEDGES);
      node_finalize_kernel<8><<<nblocks, 256, 0, stream>>>(hout, den, bias, NNODES);
    }

    float* t = hin; hin = hout; hout = t;
  }

  // hin now holds layer-4 output (N x 64).
  head_kernel<<<(NNODES + 255) / 256, 256, 0, stream>>>(
      hin, lin1_w, lin1_b, lin2_w, lin2_b, (float*)d_out, NNODES);
}
